// NeuralLensDistortion_16535624090079
// MI455X (gfx1250) — matrix operand order repaired and checked by reading, not compiled, not run on hardware
//
#include <hip/hip_runtime.h>

typedef _Float16 v16h __attribute__((ext_vector_type(16)));
typedef _Float16 v8h  __attribute__((ext_vector_type(8)));
typedef float    v8f  __attribute__((ext_vector_type(8)));

#define HDIM 256
#define MT   256        // rows per workgroup (8 waves x 32 rows each)
#define HPAD 264        // padded LDS row stride (halves) -> 528B, 16B-aligned, conflict-light
#define WPAD 264
#define NBLK 2
#define NLAY 4

union HV { v16h v; struct { v8h lo; v8h hi; } p; };

// Transpose + convert w_hid f32 [blk][l][k][j] -> f16 wT [blk][l][j][k] in workspace.
__global__ __launch_bounds__(256) void prep_weights(const float* __restrict__ w_hid,
                                                    _Float16* __restrict__ wT) {
  int idx = blockIdx.x * 256 + threadIdx.x;  // 2*4*256*256 = 524288 elements
  int k = idx & 255;
  int j = (idx >> 8) & 255;
  int m = idx >> 16;                         // blk*NLAY + l
  wT[idx] = (_Float16)w_hid[(m * 256 + k) * 256 + j];
}

__global__ __launch_bounds__(256) void mlp_kernel(
    const float* __restrict__ uv,
    const float* __restrict__ w_in, const float* __restrict__ b_in,
    const _Float16* __restrict__ wT, const float* __restrict__ b_hid,
    const float* __restrict__ w_out, const float* __restrict__ b_out,
    float* __restrict__ out) {
  __shared__ __align__(16) _Float16 hbuf[MT * HPAD];    // activations, f16, wave-private rows
  __shared__ __align__(16) _Float16 wbuf[HDIM * WPAD];  // current layer W^T, f16
  __shared__ float xbuf[MT * 2];                        // residual state x

  const int tid  = threadIdx.x;
  const int wave = tid >> 5;
  const int lane = tid & 31;
  const int g    = lane >> 4;   // half-wave group
  const int ml   = lane & 15;
  const size_t rowBase = (size_t)blockIdx.x * MT;

  for (int i = tid; i < MT * 2; i += 256)
    xbuf[i] = uv[rowBase * 2 + i];
  __syncthreads();

  for (int blk = 0; blk < NBLK; ++blk) {
    // ---------- input layer: Lin(2,256) + ELU, VALU (negligible FLOPs) ----------
    {
      const int j   = tid;  // one column per thread
      const float w0 = w_in[(blk * 2 + 0) * HDIM + j];
      const float w1 = w_in[(blk * 2 + 1) * HDIM + j];
      const float bb = b_in[blk * HDIM + j];
      #pragma unroll 4
      for (int r = 0; r < MT; ++r) {
        float v = fmaf(xbuf[r * 2], w0, fmaf(xbuf[r * 2 + 1], w1, bb));
        v = v > 0.0f ? v : (__expf(v) - 1.0f);
        hbuf[r * HPAD + j] = (_Float16)v;
      }
    }

    // ---------- hidden layers: 4x [256,256]@[256,256] via WMMA ----------
    for (int l = 0; l < NLAY; ++l) {
      __syncthreads();  // prev-layer wbuf readers done; input-layer hbuf visible
      const _Float16* wg  = wT + (size_t)(blk * NLAY + l) * HDIM * HDIM;
      const uint4*    wg4 = (const uint4*)wg;
      #pragma unroll 4
      for (int c = tid; c < HDIM * 32; c += 256) {       // 8192 x 16B chunks
        int j = c >> 5, w = c & 31;
        *(uint4*)&wbuf[j * WPAD + w * 8] = wg4[c];
      }
      if (l + 1 < NLAY) {  // warm next layer's weights into cache
        __builtin_prefetch((const char*)(wg + HDIM * HDIM) + tid * 512, 0, 1);
      }
      __syncthreads();

      const int rowOff = wave * 32;   // two 16-row M tiles per wave
      // Preload ALL A fragments for both M tiles (enables in-place update + B reuse).
      // ISA A layout: elem e of lane L -> K = t*32 + g*8 + e (e<8), t*32+16+g*8+(e-8) (e>=8)
      HV a0[8], a1[8];
      #pragma unroll
      for (int t = 0; t < 8; ++t) {
        const _Float16* b0 = &hbuf[(rowOff + ml) * HPAD + t * 32 + g * 8];
        const _Float16* b1 = &hbuf[(rowOff + 16 + ml) * HPAD + t * 32 + g * 8];
        a0[t].p.lo = *(const v8h*)(b0);
        a0[t].p.hi = *(const v8h*)(b0 + 16);
        a1[t].p.lo = *(const v8h*)(b1);
        a1[t].p.hi = *(const v8h*)(b1 + 16);
      }
      const float* bh = b_hid + (blk * NLAY + l) * HDIM;

      #pragma unroll 2
      for (int nt = 0; nt < 16; ++nt) {
        const int col = nt * 16 + ml;
        v8f c0 = {};
        v8f c1 = {};
        #pragma unroll
        for (int t = 0; t < 8; ++t) {
          // ISA B layout: elem e of lane L -> K = t*32 + g*8 + e (e<8), t*32+16+g*8+(e-8) (e>=8), column = L%16
          HV b;
          const _Float16* wb = &wbuf[col * WPAD + t * 32 + g * 8];
          b.p.lo = *(const v8h*)(wb);
          b.p.hi = *(const v8h*)(wb + 16);
          // B fragment reused by two WMMAs -> ~1 ds_load_b128 per wmma
          c0 = __builtin_amdgcn_wmma_f32_16x16x32_f16(false, a0[t].v, false, b.v,
                                                      (short)0, c0, false, false);
          c1 = __builtin_amdgcn_wmma_f32_16x16x32_f16(false, a1[t].v, false, b.v,
                                                      (short)0, c1, false, false);
        }
        // epilogue: bias + ELU + f16, write back in place (own rows only).
        // D layout: reg r -> row g*8+r, col = lane%16 (+tile offsets)
        const float bb = bh[col];
        #pragma unroll
        for (int r = 0; r < 8; ++r) {
          float v = c0[r] + bb;
          v = v > 0.0f ? v : (__expf(v) - 1.0f);
          hbuf[(rowOff + g * 8 + r) * HPAD + col] = (_Float16)v;
          float w = c1[r] + bb;
          w = w > 0.0f ? w : (__expf(w) - 1.0f);
          hbuf[(rowOff + 16 + g * 8 + r) * HPAD + col] = (_Float16)w;
        }
      }
    }
    __syncthreads();  // final hidden activations visible to all waves

    // ---------- output layer Lin(256,2) + residual: one thread per row ----------
    {
      const int r = tid;                       // 256 rows
      const float* wo = w_out + blk * HDIM * 2;
      const _Float16* hr = &hbuf[r * HPAD];
      float acc0 = 0.f, acc1 = 0.f;
      #pragma unroll 4
      for (int j = 0; j < HDIM; j += 8) {
        v8h hv = *(const v8h*)(hr + j);        // 16B LDS load
        #pragma unroll
        for (int q = 0; q < 8; ++q) {
          float hvq = (float)hv[q];
          float2 w2 = *(const float2*)(wo + (j + q) * 2);
          acc0 = fmaf(hvq, w2.x, acc0);
          acc1 = fmaf(hvq, w2.y, acc1);
        }
      }
      xbuf[r * 2 + 0] += acc0 + b_out[blk * 2 + 0];
      xbuf[r * 2 + 1] += acc1 + b_out[blk * 2 + 1];
    }
    __syncthreads();  // x updated before next block / store
  }

  for (int i = tid; i < MT * 2; i += 256)
    out[rowBase * 2 + i] = xbuf[i];
}

extern "C" void kernel_launch(void* const* d_in, const int* in_sizes, int n_in,
                              void* d_out, int out_size, void* d_ws, size_t ws_size,
                              hipStream_t stream) {
  const float* uv    = (const float*)d_in[0];
  const float* w_in  = (const float*)d_in[1];
  const float* b_in  = (const float*)d_in[2];
  const float* w_hid = (const float*)d_in[3];
  const float* b_hid = (const float*)d_in[4];
  const float* w_out = (const float*)d_in[5];
  const float* b_out = (const float*)d_in[6];
  float* outp = (float*)d_out;
  _Float16* wT = (_Float16*)d_ws;              // 1 MB of workspace

  const int n = in_sizes[0] / 2;               // number of points

  prep_weights<<<(NBLK * NLAY * HDIM * HDIM) / 256, 256, 0, stream>>>(w_hid, wT);
  mlp_kernel<<<n / MT, 256, 0, stream>>>(uv, w_in, b_in, wT, b_hid, w_out, b_out, outp);
}
